// TokenMergingAttention_69827578298925
// MI455X (gfx1250) — matrix operand order repaired and checked by reading, not compiled, not run on hardware
//
#include <hip/hip_runtime.h>

#define BATCH   4
#define SEQ     1024
#define EMB     1024
#define HEADS   16
#define HDIM    64
#define SIMSCALE 0.125f   // 1/sqrt(64)
#define THRESH  0.9f

typedef __attribute__((ext_vector_type(8)))  __bf16 bf16x8;
typedef __attribute__((ext_vector_type(16))) __bf16 bf16x16;
typedef __attribute__((ext_vector_type(8)))  float  f32x8;

// ---- WMMA helpers (CDNA5 16x16x32 bf16, wave32) -------------------------
// A 16x32 layout: lane l (l15 = l&15 is row M), half = l>>4:
//   VGPRs 0-3 hold K = half*8 .. half*8+7 ; VGPRs 4-7 hold K = 16+half*8 ..
// B 32x16 layout: lane l -> column N = l&15, same K split as A (half*8.., 16+half*8..)
// C/D v8f: VGPR r -> element (M = half*8 + r, N = l&15)

static __device__ __forceinline__ bf16x16 load_a16(const __bf16* row, int half) {
  bf16x8 lo = *(const bf16x8*)(row + half * 8);
  bf16x8 hi = *(const bf16x8*)(row + 16 + half * 8);
  return __builtin_shufflevector(lo, hi, 0,1,2,3,4,5,6,7,8,9,10,11,12,13,14,15);
}
static __device__ __forceinline__ bf16x16 load_b16(const __bf16* row, int half) {
  bf16x8 lo = *(const bf16x8*)(row + half * 8);
  bf16x8 hi = *(const bf16x8*)(row + 16 + half * 8);
  return __builtin_shufflevector(lo, hi, 0,1,2,3,4,5,6,7,8,9,10,11,12,13,14,15);
}
static __device__ __forceinline__ f32x8 wmma_bf16(bf16x16 a, bf16x16 b, f32x8 c) {
  return __builtin_amdgcn_wmma_f32_16x16x32_bf16(false, a, false, b, (short)0, c,
                                                 false, false);
}

// ---- K0: convert x/wq/wk/wo to bf16, batch-mean of x, zero keep ---------
__global__ void __launch_bounds__(256) k0_prep(
    const float* __restrict__ x, const float* __restrict__ wq,
    const float* __restrict__ wk, const float* __restrict__ wo,
    __bf16* __restrict__ xb, __bf16* __restrict__ wqb,
    __bf16* __restrict__ wkb, __bf16* __restrict__ wob,
    float* __restrict__ meanx, unsigned* __restrict__ keep) {
  int i = blockIdx.x * 256 + threadIdx.x;              // grid = 4M threads
  xb[i] = (__bf16)x[i];
  if (i < EMB * EMB) {
    wqb[i] = (__bf16)wq[i];
    wkb[i] = (__bf16)wk[i];
    wob[i] = (__bf16)wo[i];
    meanx[i] = 0.25f * (x[i] + x[i + 1048576] + x[i + 2097152] + x[i + 3145728]);
  }
  if (i < SEQ) keep[i] = 0u;
}

// ---- K1: Q/K projection, NT GEMM, one 16x16 tile each per wave ----------
__global__ void __launch_bounds__(256) k1_qkproj(
    const __bf16* __restrict__ xb, const __bf16* __restrict__ wqb,
    const __bf16* __restrict__ wkb, const float* __restrict__ bq,
    const float* __restrict__ bk, __bf16* __restrict__ qout,
    __bf16* __restrict__ kout) {
  int wid  = blockIdx.x * 8 + (threadIdx.x >> 5);      // 16384 wave tiles
  int lane = threadIdx.x & 31, l15 = lane & 15, half = lane >> 4;
  int nt = wid & 63, mt = wid >> 6;
  int row0 = mt * 16;                                  // b*SEQ + s
  int b = row0 >> 10, s0 = row0 & 1023;
  int e0 = nt * 16, h = e0 >> 6, d0 = e0 & 63;
  const __bf16* arow  = xb  + (size_t)(row0 + l15) * EMB;
  const __bf16* bqrow = wqb + (size_t)(e0 + l15) * EMB;
  const __bf16* bkrow = wkb + (size_t)(e0 + l15) * EMB;
  f32x8 cq = {}, ck = {};
  for (int k0 = 0; k0 < EMB; k0 += 32) {
    bf16x16 a = load_a16(arow + k0, half);
    cq = wmma_bf16(a, load_b16(bqrow + k0, half), cq);
    ck = wmma_bf16(a, load_b16(bkrow + k0, half), ck);
  }
  float bqv = bq[e0 + l15], bkv = bk[e0 + l15];
  size_t ob = ((size_t)(b * HEADS + h) * SEQ + s0) * HDIM + d0 + l15;
#pragma unroll
  for (int r = 0; r < 8; ++r) {
    int m = half * 8 + r;
    qout[ob + (size_t)m * HDIM] = (__bf16)(cq[r] + bqv);
    kout[ob + (size_t)m * HDIM] = (__bf16)(ck[r] + bkv);
  }
}

// ---- K2: sim row stats (online softmax max/sum) + keep mask -------------
__global__ void __launch_bounds__(256) k2_stats(
    const __bf16* __restrict__ qb, const __bf16* __restrict__ kbuf,
    float* __restrict__ rowmax, float* __restrict__ rowsum,
    unsigned* __restrict__ keep) {
  __shared__ float smax[8][16], ssum[8][16];
  int bid = blockIdx.x;                                // 4096 = B*H*(S/16)
  int q0 = (bid & 63) * 16, bh = bid >> 6;
  int w = threadIdx.x >> 5, lane = threadIdx.x & 31, l15 = lane & 15, half = lane >> 4;
  const __bf16* qbase = qb   + (size_t)bh * SEQ * HDIM;
  const __bf16* kbase = kbuf + (size_t)bh * SEQ * HDIM;
  const __bf16* arow  = qbase + (size_t)(q0 + l15) * HDIM;
  bf16x16 a0 = load_a16(arow, half), a1 = load_a16(arow + 32, half);
  float rmx[8], rsm[8];
#pragma unroll
  for (int r = 0; r < 8; ++r) { rmx[r] = -1e30f; rsm[r] = 0.f; }
  unsigned kf = 0;
  for (int t = 0; t < 8; ++t) {                        // wave covers 128 keys
    int k0 = w * 128 + t * 16;
    const __bf16* brow = kbase + (size_t)(k0 + l15) * HDIM;
    f32x8 c = {};
    c = wmma_bf16(a0, load_b16(brow, half), c);
    c = wmma_bf16(a1, load_b16(brow + 32, half), c);
#pragma unroll
    for (int r = 0; r < 8; ++r) {
      float s = c[r] * SIMSCALE;
      if (s > THRESH) kf |= 1u << r;
      float tm = s;                                    // row max over 16 lanes
      tm = fmaxf(tm, __shfl_xor(tm, 1, 32));
      tm = fmaxf(tm, __shfl_xor(tm, 2, 32));
      tm = fmaxf(tm, __shfl_xor(tm, 4, 32));
      tm = fmaxf(tm, __shfl_xor(tm, 8, 32));
      float nm = fmaxf(rmx[r], tm);
      float e = __expf(s - nm);
      e += __shfl_xor(e, 1, 32);
      e += __shfl_xor(e, 2, 32);
      e += __shfl_xor(e, 4, 32);
      e += __shfl_xor(e, 8, 32);
      rsm[r] = rsm[r] * __expf(rmx[r] - nm) + e;
      rmx[r] = nm;
    }
  }
#pragma unroll
  for (int r = 0; r < 8; ++r) {
    unsigned long long mm = __ballot((kf >> r) & 1);
    if (lane == 0  && (mm & 0xFFFFull))       atomicOr(&keep[q0 + r], 1u);
    if (lane == 16 && (mm & 0xFFFF0000ull))   atomicOr(&keep[q0 + 8 + r], 1u);
  }
  if (l15 == 0) {
#pragma unroll
    for (int r = 0; r < 8; ++r) {
      smax[w][half * 8 + r] = rmx[r];
      ssum[w][half * 8 + r] = rsm[r];
    }
  }
  __syncthreads();
  if (threadIdx.x < 16) {
    int m = threadIdx.x;
    float gm = -1e30f;
    for (int w2 = 0; w2 < 8; ++w2) gm = fmaxf(gm, smax[w2][m]);
    float gs = 0.f;
    for (int w2 = 0; w2 < 8; ++w2) gs += ssum[w2][m] * __expf(smax[w2][m] - gm);
    size_t ridx = (size_t)bh * SEQ + q0 + m;
    rowmax[ridx] = gm;
    rowsum[ridx] = gs;
  }
}

// ---- K3a: masked transpose  mT[k][m] = keep[m] * meanx[m][k] ------------
__global__ void __launch_bounds__(256) k3a_mask(
    const float* __restrict__ meanx, const unsigned* __restrict__ keep,
    __bf16* __restrict__ mT) {
  int i = blockIdx.x * 256 + threadIdx.x;              // 1M threads
  int k = i >> 10, m = i & 1023;
  float v = keep[m] ? meanx[(size_t)m * 1024 + k] : 0.f;
  mT[i] = (__bf16)v;
}

// ---- K3b: W2T[e][k] = sum_m wo[e][m] * mT[k][m]  (NT GEMM) --------------
__global__ void __launch_bounds__(256) k3b_w2(
    const __bf16* __restrict__ wob, const __bf16* __restrict__ mT,
    __bf16* __restrict__ w2t) {
  int wid  = blockIdx.x * 8 + (threadIdx.x >> 5);      // 4096 wave tiles
  int lane = threadIdx.x & 31, l15 = lane & 15, half = lane >> 4;
  int e0 = (wid >> 6) * 16, k0 = (wid & 63) * 16;
  const __bf16* arow = wob + (size_t)(e0 + l15) * 1024;
  const __bf16* brow = mT  + (size_t)(k0 + l15) * 1024;
  f32x8 c = {};
  for (int m0 = 0; m0 < 1024; m0 += 32)
    c = wmma_bf16(load_a16(arow + m0, half), load_b16(brow + m0, half), c);
  __bf16* ob = w2t + (size_t)e0 * 1024 + k0 + l15;
#pragma unroll
  for (int r = 0; r < 8; ++r)
    ob[(size_t)(half * 8 + r) * 1024] = (__bf16)c[r];
}

// ---- K4: fused  out = softmax(sim) @ W2 + bo  (flash-style) -------------
// Block = (b,h, 32-row q-block). 16 waves: each produces one 32x32 p-chunk
// per 512-k phase into LDS (8 sim WMMAs), then all waves consume all 16
// chunks for their own 64-column e-strip, reusing each W2T B-tile across
// both q-tiles. 32 q-rows/WG halves W2T L2 traffic vs 16 (8 GB -> 4 GB).
__global__ void __launch_bounds__(512) k4_out(
    const __bf16* __restrict__ qb, const __bf16* __restrict__ kbuf,
    const __bf16* __restrict__ w2t, const float* __restrict__ rowmax,
    const float* __restrict__ rowsum, const float* __restrict__ bo,
    float* __restrict__ out) {
  __shared__ __attribute__((aligned(16))) __bf16 pbuf[16][32][32];  // 32 KB
  int bid = blockIdx.x;                                // 2048 = B*H*(S/32)
  int q0 = (bid & 31) * 32, bh = bid >> 5;
  int w = threadIdx.x >> 5, lane = threadIdx.x & 31, l15 = lane & 15, half = lane >> 4;
  int ebase = w * 64;                                  // 16 waves x 64 cols
  const __bf16* kbase = kbuf + (size_t)bh * SEQ * HDIM;
  bf16x16 aq[2][2];
#pragma unroll
  for (int qt = 0; qt < 2; ++qt) {
    const __bf16* qrow = qb + ((size_t)bh * SEQ + q0 + qt * 16 + l15) * HDIM;
    aq[qt][0] = load_a16(qrow, half);
    aq[qt][1] = load_a16(qrow + 32, half);
  }
  float rmx[2][8], rin[2][8];
#pragma unroll
  for (int qt = 0; qt < 2; ++qt)
#pragma unroll
    for (int r = 0; r < 8; ++r) {
      size_t ridx = (size_t)bh * SEQ + q0 + qt * 16 + half * 8 + r;
      rmx[qt][r] = rowmax[ridx];
      rin[qt][r] = 1.0f / rowsum[ridx];
    }
  f32x8 acc[2][4] = {};
  for (int kb = 0; kb < SEQ; kb += 512) {              // 2 phases
    int k0 = kb + w * 32;
#pragma unroll
    for (int t = 0; t < 2; ++t) {                      // produce p-chunk
      const __bf16* brow = kbase + (size_t)(k0 + t * 16 + l15) * HDIM;
      bf16x16 b0 = load_b16(brow, half);
      bf16x16 b1 = load_b16(brow + 32, half);
#pragma unroll
      for (int qt = 0; qt < 2; ++qt) {
        f32x8 c = {};
        c = wmma_bf16(aq[qt][0], b0, c);
        c = wmma_bf16(aq[qt][1], b1, c);
#pragma unroll
        for (int r = 0; r < 8; ++r)
          pbuf[w][qt * 16 + half * 8 + r][t * 16 + l15] =
              (__bf16)__expf(c[r] * SIMSCALE - rmx[qt][r]);
      }
    }
    __syncthreads();
#pragma unroll
    for (int wc = 0; wc < 16; ++wc) {                  // consume all chunks
      bf16x16 pa0 = load_a16(&pbuf[wc][l15][0], half);       // q-tile 0
      bf16x16 pa1 = load_a16(&pbuf[wc][16 + l15][0], half);  // q-tile 1
      int kg = kb + wc * 32;
#pragma unroll
      for (int t = 0; t < 4; ++t) {
        const __bf16* brow = w2t + (size_t)(ebase + t * 16 + l15) * 1024 + kg;
        if (kb == 0)                                   // prefetch phase 2 rows
          __builtin_prefetch((const void*)(brow + 512), 0, 3);
        bf16x16 bmat = load_b16(brow, half);
        acc[0][t] = wmma_bf16(pa0, bmat, acc[0][t]);
        acc[1][t] = wmma_bf16(pa1, bmat, acc[1][t]);
      }
    }
    __syncthreads();                                   // protect pbuf reuse
  }
  float* obase = out + ((size_t)bh * SEQ + q0) * EMB;
#pragma unroll
  for (int qt = 0; qt < 2; ++qt)
#pragma unroll
    for (int t = 0; t < 4; ++t) {
      int e = ebase + t * 16 + l15;
      float bov = bo[e];
#pragma unroll
      for (int r = 0; r < 8; ++r)
        obase[(size_t)(qt * 16 + half * 8 + r) * EMB + e] =
            acc[qt][t][r] * rin[qt][r] + bov;
    }
}

// ---- host launcher -------------------------------------------------------
extern "C" void kernel_launch(void* const* d_in, const int* in_sizes, int n_in,
                              void* d_out, int out_size, void* d_ws, size_t ws_size,
                              hipStream_t stream) {
  (void)in_sizes; (void)n_in; (void)out_size; (void)ws_size;
  const float* x  = (const float*)d_in[0];
  const float* wq = (const float*)d_in[1];
  const float* bq = (const float*)d_in[2];
  const float* wk = (const float*)d_in[3];
  const float* bk = (const float*)d_in[4];
  const float* wo = (const float*)d_in[7];
  const float* bo = (const float*)d_in[8];
  float* out = (float*)d_out;

  char* p = (char*)d_ws;                               // ~40 MB total
  auto alloc = [&](size_t bytes) {
    char* r = p;
    p += (bytes + 255) & ~(size_t)255;
    return r;
  };
  __bf16*   xb    = (__bf16*)alloc(sizeof(__bf16) * BATCH * SEQ * EMB);   // 8 MB
  __bf16*   wqb   = (__bf16*)alloc(sizeof(__bf16) * EMB * EMB);           // 2 MB
  __bf16*   wkb   = (__bf16*)alloc(sizeof(__bf16) * EMB * EMB);           // 2 MB
  __bf16*   wob   = (__bf16*)alloc(sizeof(__bf16) * EMB * EMB);           // 2 MB
  float*    meanx = (float*) alloc(sizeof(float)  * SEQ * EMB);           // 4 MB
  __bf16*   qbuf  = (__bf16*)alloc(sizeof(__bf16) * BATCH * SEQ * EMB);   // 8 MB
  __bf16*   kbuf  = (__bf16*)alloc(sizeof(__bf16) * BATCH * SEQ * EMB);   // 8 MB
  __bf16*   mT    = (__bf16*)alloc(sizeof(__bf16) * SEQ * EMB);           // 2 MB
  __bf16*   w2t   = (__bf16*)alloc(sizeof(__bf16) * EMB * SEQ);           // 2 MB
  float*    rmax  = (float*) alloc(sizeof(float)  * BATCH * HEADS * SEQ); // 256 KB
  float*    rsum  = (float*) alloc(sizeof(float)  * BATCH * HEADS * SEQ); // 256 KB
  unsigned* keep  = (unsigned*)alloc(sizeof(unsigned) * SEQ);             // 4 KB

  k0_prep  <<<16384, 256, 0, stream>>>(x, wq, wk, wo, xb, wqb, wkb, wob, meanx, keep);
  k1_qkproj<<<2048,  256, 0, stream>>>(xb, wqb, wkb, bq, bk, qbuf, kbuf);
  k2_stats <<<4096,  256, 0, stream>>>(qbuf, kbuf, rmax, rsum, keep);
  k3a_mask <<<4096,  256, 0, stream>>>(meanx, keep, mT);
  k3b_w2   <<<512,   256, 0, stream>>>(wob, mT, w2t);
  k4_out   <<<2048,  512, 0, stream>>>(qbuf, kbuf, w2t, rmax, rsum, bo, out);
}
